// Int8Linear_39943195852889
// MI455X (gfx1250) — compile-verified, matrix operand order checked by reading, not run on hardware
//
#include <hip/hip_runtime.h>
#include <stdint.h>

typedef __attribute__((ext_vector_type(16))) _Float16 v16h;
typedef __attribute__((ext_vector_type(8)))  _Float16 v8h;
typedef __attribute__((ext_vector_type(2)))  _Float16 v2h;
typedef __attribute__((ext_vector_type(8)))  float    v8f;
typedef __attribute__((ext_vector_type(4)))  float    f32x4;
typedef __attribute__((ext_vector_type(4)))  int      i32x4;

#if __has_builtin(__builtin_amdgcn_global_load_async_to_lds_b128)
#define HAVE_ASYNC_LDS 1
#else
#define HAVE_ASYNC_LDS 0
#endif

namespace {
constexpr int M_TOT = 8192;    // B * S
constexpr int N_TOT = 11008;   // OUT_F
constexpr int K_TOT = 4096;    // IN_F
constexpr int BM = 256;
constexpr int BN = 128;
constexpr int BK = 32;
constexpr int LDA = BK + 8;    // padded LDS stride in halves (80B rows, conflict-free b128)
constexpr int KT  = K_TOT / BK;
constexpr int BUF_STRIDE  = (BM + BN) * LDA;       // halves per double-buffer slot
constexpr int B_OFFSET    = BM * LDA;              // B tile offset within a slot
constexpr int SMEM_HALVES = 2 * BUF_STRIDE;        // 30720 halves = 60 KB
}

#if HAVE_ASYNC_LDS
typedef __attribute__((address_space(1))) i32x4 g_i32x4;   // global (device) AS
typedef __attribute__((address_space(3))) i32x4 l_i32x4;   // LDS AS
// Memory -> LDS without VGPR staging; tracked by ASYNCcnt.
__device__ inline void async_copy_b128(const void* g, void* l)
{
    __builtin_amdgcn_global_load_async_to_lds_b128((g_i32x4*)g, (l_i32x4*)l, 0, 0);
}
__device__ inline void wait_async0()
{
#if __has_builtin(__builtin_amdgcn_s_wait_asynccnt)
    __builtin_amdgcn_s_wait_asynccnt(0);
#else
    asm volatile("s_wait_asynccnt 0x0" ::: "memory");
#endif
}
#endif

// Exact int8 -> fp16 via exponent-bias trick:
//   u8 = s8 ^ 0x80;  as_half(0x6400 | u8) == 1024 + u8;  subtract 1152 => s8.
// 1 xor + 2 v_perm_b32 + 2 v_pk_add_f16 per input dword (1.25 VALU/byte).
__device__ inline void cvt_dword_i8_to_f16(unsigned w, unsigned& lo, unsigned& hi)
{
    const unsigned e64 = 0x64646464u;
    unsigned x = w ^ 0x80808080u;
    unsigned plo = __builtin_amdgcn_perm(e64, x, 0x04010400u); // [b0',0x64,b1',0x64]
    unsigned phi = __builtin_amdgcn_perm(e64, x, 0x04030402u); // [b2',0x64,b3',0x64]
    union { unsigned u; v2h h; } a, b;
    const v2h bias = { (_Float16)1152.0f, (_Float16)1152.0f };
    a.u = plo; b.u = phi;
    a.h = a.h - bias;
    b.h = b.h - bias;
    lo = a.u; hi = b.u;
}

__global__ __launch_bounds__(256)
void int8_linear_wmma_kernel(const _Float16* __restrict__ X,
                             const int8_t*   __restrict__ W,
                             const _Float16* __restrict__ SCL,
                             const _Float16* __restrict__ BIA,
                             _Float16*       __restrict__ Y)
{
    extern __shared__ _Float16 smem[];

    const int tid   = threadIdx.x;
    const int lane  = tid & 31;
    const int wave  = tid >> 5;
    const int waveM = wave >> 2;   // 0..1 -> 128-row slab
    const int waveN = wave & 3;    // 0..3 -> 32-col slab
    const int rl    = lane & 15;   // row-within-tile lane
    const int hsel  = lane >> 4;   // K-half select (WMMA 16-bit A/B layout)

    const int m0 = blockIdx.y * BM;
    const int n0 = blockIdx.x * BN;

    const _Float16* gA = X + (size_t)m0 * K_TOT;
    const int8_t*   gB = W + (size_t)n0 * K_TOT;

    v8f acc[8][2];
    {
        v8f z = {0.f, 0.f, 0.f, 0.f, 0.f, 0.f, 0.f, 0.f};
        #pragma unroll
        for (int mi = 0; mi < 8; ++mi)
            #pragma unroll
            for (int ni = 0; ni < 2; ++ni)
                acc[mi][ni] = z;
    }

    // Per-thread staging geometry (hoisted: one add per stream per iteration).
    int ar_[4], ac_[4];
    #pragma unroll
    for (int j = 0; j < 4; ++j) {
        int c = tid + j * 256;              // 1024 chunks of 8 halves (A: 256x32)
        ar_[j] = c >> 2; ac_[j] = (c & 3) * 8;
    }
    const int br_ = tid >> 1, bc_ = (tid & 1) * 16;   // 256 chunks of 16 bytes (B: 128x32)

    const _Float16* aP[4];
    #pragma unroll
    for (int j = 0; j < 4; ++j) aP[j] = gA + (size_t)ar_[j] * K_TOT + ac_[j];
    const int8_t* bP = gB + (size_t)br_ * K_TOT + bc_;

#if !HAVE_ASYNC_LDS
    f32x4 regA[4];   // manual A staging fallback
#endif
    i32x4 regB;      // staged W: 16 int8 per thread

    // Issue phase: A goes memory->LDS directly (ASYNCcnt), B into registers.
    auto stage_issue = [&](int buf, int kt) {
#if HAVE_ASYNC_LDS
        _Float16* sA = smem + buf * BUF_STRIDE;
        #pragma unroll
        for (int j = 0; j < 4; ++j)
            async_copy_b128(aP[j] + (size_t)kt * BK, sA + ar_[j] * LDA + ac_[j]);
#else
        (void)buf;
        #pragma unroll
        for (int j = 0; j < 4; ++j)
            regA[j] = *(const f32x4*)(aP[j] + (size_t)kt * BK);
#endif
        regB = *(const i32x4*)(bP + (size_t)kt * BK);
    };

    // Commit phase: dequantize B into LDS, land async A, then barrier outside.
    auto stage_commit = [&](int buf) {
        _Float16* sA = smem + buf * BUF_STRIDE;
        _Float16* sB = sA + B_OFFSET;
#if !HAVE_ASYNC_LDS
        #pragma unroll
        for (int j = 0; j < 4; ++j)
            *(f32x4*)(sA + ar_[j] * LDA + ac_[j]) = regA[j];
#endif
        {
            const unsigned* wsrc = (const unsigned*)&regB;
            union { unsigned u[4]; i32x4 v; } o0, o1;
            cvt_dword_i8_to_f16(wsrc[0], o0.u[0], o0.u[1]);
            cvt_dword_i8_to_f16(wsrc[1], o0.u[2], o0.u[3]);
            cvt_dword_i8_to_f16(wsrc[2], o1.u[0], o1.u[1]);
            cvt_dword_i8_to_f16(wsrc[3], o1.u[2], o1.u[3]);
            *(i32x4*)(sB + br_ * LDA + bc_)     = o0.v;
            *(i32x4*)(sB + br_ * LDA + bc_ + 8) = o1.v;
        }
#if HAVE_ASYNC_LDS
        wait_async0();    // this wave's async A copies have landed in LDS
#endif
    };

    auto compute = [&](int buf) {
        const _Float16* sA = smem + buf * BUF_STRIDE;
        const _Float16* sB = sA + B_OFFSET;
        v16h bfr[2];   // B fragments reused by all 8 M tiles
        #pragma unroll
        for (int ni = 0; ni < 2; ++ni) {
            const _Float16* p = sB + (waveN * 32 + ni * 16 + rl) * LDA + hsel * 8;
            v8h lo = *(const v8h*)p;
            v8h hi = *(const v8h*)(p + 16);
            #pragma unroll
            for (int i = 0; i < 8; ++i) { bfr[ni][i] = lo[i]; bfr[ni][i + 8] = hi[i]; }
        }
        #pragma unroll
        for (int mi = 0; mi < 8; ++mi) {
            const _Float16* p = sA + (waveM * 128 + mi * 16 + rl) * LDA + hsel * 8;
            v8h lo = *(const v8h*)p;
            v8h hi = *(const v8h*)(p + 16);
            v16h afr;
            #pragma unroll
            for (int i = 0; i < 8; ++i) { afr[i] = lo[i]; afr[i + 8] = hi[i]; }
            #pragma unroll
            for (int ni = 0; ni < 2; ++ni)
                acc[mi][ni] = __builtin_amdgcn_wmma_f32_16x16x32_f16(
                    false, afr, false, bfr[ni],
                    (short)0, acc[mi][ni], false, false);
        }
    };

    stage_issue(0, 0);
    stage_commit(0);
    __syncthreads();

    for (int kt = 0; kt < KT; ++kt) {
        const int cur = kt & 1;
        if (kt + 1 < KT) stage_issue(cur ^ 1, kt + 1);
        if (kt + 4 < KT) {
            __builtin_prefetch(aP[0] + (size_t)(kt + 4) * BK, 0, 0);
            __builtin_prefetch(bP + (size_t)(kt + 4) * BK, 0, 0);
        }
        compute(cur);
        if (kt + 1 < KT) {
            stage_commit(cur ^ 1);   // writes the buffer no one is reading
            __syncthreads();
        }
    }

    // Epilogue: y = acc * scale[n] + bias[n]
    #pragma unroll
    for (int ni = 0; ni < 2; ++ni) {
        const int n  = n0 + waveN * 32 + ni * 16 + rl;
        const float sf = (float)SCL[n];
        const float bf = (float)BIA[n];
        #pragma unroll
        for (int mi = 0; mi < 8; ++mi) {
            const int mb = m0 + waveM * 128 + mi * 16 + hsel * 8;  // C layout: VGPR i -> M=i(+8)
            #pragma unroll
            for (int i = 0; i < 8; ++i) {
                Y[(size_t)(mb + i) * N_TOT + n] = (_Float16)(acc[mi][ni][i] * sf + bf);
            }
        }
    }
}

extern "C" void kernel_launch(void* const* d_in, const int* in_sizes, int n_in,
                              void* d_out, int out_size, void* d_ws, size_t ws_size,
                              hipStream_t stream) {
    (void)in_sizes; (void)n_in; (void)out_size; (void)d_ws; (void)ws_size;
    const _Float16* x  = (const _Float16*)d_in[0];
    const int8_t*   w  = (const int8_t*)d_in[1];
    const _Float16* sc = (const _Float16*)d_in[2];
    const _Float16* bi = (const _Float16*)d_in[3];
    _Float16*       y  = (_Float16*)d_out;

    dim3 grid(N_TOT / BN, M_TOT / BM);   // 86 x 32 blocks
    dim3 block(256);                     // 8 wave32 waves
    size_t shmem = (size_t)SMEM_HALVES * sizeof(_Float16);   // 60 KB, double-buffered
    int8_linear_wmma_kernel<<<grid, block, shmem, stream>>>(x, w, sc, bi, y);
}
